// Affordance_46961172414788
// MI455X (gfx1250) — matrix operand order repaired and checked by reading, not compiled, not run on hardware
//
#include <hip/hip_runtime.h>
#include <hip/hip_bf16.h>

// ---------- types ----------
typedef __bf16 bf16_t;
typedef __attribute__((ext_vector_type(16))) bf16_t v16bf;
typedef __attribute__((ext_vector_type(8)))  float  v8f;
typedef unsigned short u16;
typedef __attribute__((ext_vector_type(8)))  u16    u16x8;

union FragBF { v16bf v; u16x8 h[2]; };

__device__ __forceinline__ u16 f2bf(float f) {
    unsigned u = __builtin_bit_cast(unsigned, f);
    unsigned r = u + 0x7FFFu + ((u >> 16) & 1u);   // round-to-nearest-even
    return (u16)(r >> 16);
}
__device__ __forceinline__ float bf2f(u16 h) {
    unsigned u = ((unsigned)h) << 16;
    return __builtin_bit_cast(float, u);
}

// ---------- weight / misc conversion ----------
__global__ void k_convw(const float* __restrict__ w, u16* __restrict__ dst,
                        int N, int K, int Kp) {
    int t = blockIdx.x * blockDim.x + threadIdx.x;
    if (t >= N * Kp) return;
    int n = t / Kp, k = t % Kp;
    dst[t] = (k < K) ? f2bf(w[(size_t)n * K + k]) : (u16)0;
}

__global__ void k_cvt(const float* __restrict__ s, u16* __restrict__ d, int n) {
    int t = blockIdx.x * blockDim.x + threadIdx.x;
    if (t < n) d[t] = f2bf(s[t]);
}

// ---------- farthest point sampling: one block per batch ----------
__global__ __launch_bounds__(256)
void k_fps(const float* __restrict__ xyz, int n, int npoint,
           int* __restrict__ fidx, float* __restrict__ nxyz) {
    __shared__ float dist[4096];
    __shared__ float rv[256];
    __shared__ int   ri[256];
    __shared__ float lx, ly, lz;
    int b = blockIdx.x, t = threadIdx.x;
    const float* p  = xyz  + (size_t)b * n * 3;
    int*         fo = fidx + (size_t)b * npoint;
    float*       no = nxyz + (size_t)b * npoint * 3;
    for (int i = t; i < n; i += 256) dist[i] = 1e10f;
    if (t == 0) { lx = p[0]; ly = p[1]; lz = p[2];
                  fo[0] = 0; no[0] = lx; no[1] = ly; no[2] = lz; }
    __syncthreads();
    for (int it = 1; it < npoint; ++it) {
        float cx = lx, cy = ly, cz = lz;
        float bv = -1.0f; int bi = 0x7fffffff;
        for (int i = t; i < n; i += 256) {
            float dx = p[i*3]-cx, dy = p[i*3+1]-cy, dz = p[i*3+2]-cz;
            float d = dx*dx + dy*dy + dz*dz;
            float nd = dist[i]; nd = d < nd ? d : nd; dist[i] = nd;
            if (nd > bv || (nd == bv && i < bi)) { bv = nd; bi = i; }
        }
        rv[t] = bv; ri[t] = bi;
        __syncthreads();
        for (int s = 128; s > 0; s >>= 1) {
            if (t < s) {
                float ov = rv[t+s]; int oi = ri[t+s];
                if (ov > rv[t] || (ov == rv[t] && oi < ri[t])) { rv[t] = ov; ri[t] = oi; }
            }
            __syncthreads();
        }
        if (t == 0) {
            int s = ri[0];
            lx = p[s*3]; ly = p[s*3+1]; lz = p[s*3+2];
            fo[it] = s; no[it*3] = lx; no[it*3+1] = ly; no[it*3+2] = lz;
        }
        __syncthreads();
    }
}

// ---------- ball query: one thread per center, ordered scan ----------
__global__ void k_bq(const float* __restrict__ xyz, const float* __restrict__ nxyz,
                     int n, int npoint, float r2, int* __restrict__ idx) {
    int t = blockIdx.x * blockDim.x + threadIdx.x;
    if (t >= 16 * npoint) return;
    int b = t / npoint;
    const float* p   = xyz  + (size_t)b * n * 3;
    const float* ctr = nxyz + (size_t)t * 3;
    float cx = ctr[0], cy = ctr[1], cz = ctr[2];
    int* o = idx + (size_t)t * 32;
    int cnt = 0, first = 0;
    for (int i = 0; i < n && cnt < 32; ++i) {
        float dx = p[i*3]-cx, dy = p[i*3+1]-cy, dz = p[i*3+2]-cz;
        if (dx*dx + dy*dy + dz*dz < r2) {
            if (cnt == 0) first = i;
            o[cnt++] = i;
        }
    }
    for (; cnt < 32; ++cnt) o[cnt] = first;
}

// ---------- build grouped SA input rows: [g_xyz(3) | feats(Cf) | pad] ----------
__global__ void k_group(const float* __restrict__ xyz, const float* __restrict__ nxyz,
                        const u16* __restrict__ feats, int Cf,
                        const int* __restrict__ idx, int n, int npoint,
                        int Kp, u16* __restrict__ X) {
    int r = blockIdx.x * blockDim.x + threadIdx.x;
    int rows = 16 * npoint * 32;
    if (r >= rows) return;
    int c = (r >> 5) % npoint;
    int b = r / (npoint * 32);
    int pid = idx[r];
    const float* p   = xyz  + ((size_t)b * n + pid) * 3;
    const float* ctr = nxyz + ((size_t)b * npoint + c) * 3;
    u16* xr = X + (size_t)r * Kp;
    xr[0] = f2bf(p[0] - ctr[0]);
    xr[1] = f2bf(p[1] - ctr[1]);
    xr[2] = f2bf(p[2] - ctr[2]);
    const u16* f = feats + ((size_t)b * n + pid) * Cf;
    for (int j = 0; j < Cf; ++j) xr[3 + j] = f[j];
    for (int j = 3 + Cf; j < Kp; ++j) xr[j] = 0;
}

// ---------- WMMA GEMM: out = act((X @ W^T) * scale + bias) ----------
// X: (M,Kp) bf16 row-major, W: (N,Kp) bf16 row-major.
// One wave computes a 32x32 output block (2x2 WMMA subtiles) with
// double-buffered fragments so loads for stage k+1 overlap WMMAs of stage k.
__global__ __launch_bounds__(256)
void k_gemm(const u16* __restrict__ X, int ldx,
            const u16* __restrict__ W, int ldw,
            int M, int N, int K,
            const float* __restrict__ scale, const float* __restrict__ bias,
            int act, void* __restrict__ out, int ldo, int outbf, int coloff) {
    int wave = (blockIdx.x * blockDim.x + threadIdx.x) >> 5;
    int lane = threadIdx.x & 31;
    int nt2 = N >> 5;                       // 32-wide N blocks (all N are mult of 32)
    int mt2 = (M + 31) >> 5;                // 32-high M blocks (M=16 handled below)
    long tiles = (long)mt2 * nt2;
    if (wave >= tiles) return;
    int mt = wave / nt2, nt = wave % nt2;
    int half = lane >> 4, l = lane & 15;

    bool hasM1 = (mt * 32 + 16) < M;
    const u16* xr0 = X + (size_t)(mt * 32 + l) * ldx;
    const u16* xr1 = X + (size_t)(mt * 32 + (hasM1 ? 16 : 0) + l) * ldx;
    const u16* wr0 = W + (size_t)(nt * 32 + l) * ldw;
    const u16* wr1 = W + (size_t)(nt * 32 + 16 + l) * ldw;

    v8f acc00 = {}, acc01 = {}, acc10 = {}, acc11 = {};
    FragBF a0c, a1c, b0c, b1c, a0n, a1n, b0n, b1n;

    // A (16x32 bf16): lanes<16 hold K{0..7,16..23}, lanes>=16 hold K{8..15,24..31}
    // B (32x16 bf16) = W^T: lanes<16 hold W[col n, K{0..7,16..23}], lanes>=16 hold K{8..15,24..31}
#define LOAD_STAGE(A0, A1, B0, B1, KC)                                      \
    do {                                                                    \
        (A0).h[0] = *(const u16x8*)(xr0 + (KC) + half * 8);                 \
        (A0).h[1] = *(const u16x8*)(xr0 + (KC) + half * 8 + 16);            \
        (A1).h[0] = *(const u16x8*)(xr1 + (KC) + half * 8);                 \
        (A1).h[1] = *(const u16x8*)(xr1 + (KC) + half * 8 + 16);            \
        (B0).h[0] = *(const u16x8*)(wr0 + (KC) + half * 8);                 \
        (B0).h[1] = *(const u16x8*)(wr0 + (KC) + half * 8 + 16);            \
        (B1).h[0] = *(const u16x8*)(wr1 + (KC) + half * 8);                 \
        (B1).h[1] = *(const u16x8*)(wr1 + (KC) + half * 8 + 16);            \
    } while (0)

#define WMMA_STAGE(A0, A1, B0, B1)                                          \
    do {                                                                    \
        acc00 = __builtin_amdgcn_wmma_f32_16x16x32_bf16(                    \
                    false, (A0).v, false, (B0).v, (short)0, acc00, false, false); \
        acc01 = __builtin_amdgcn_wmma_f32_16x16x32_bf16(                    \
                    false, (A0).v, false, (B1).v, (short)0, acc01, false, false); \
        acc10 = __builtin_amdgcn_wmma_f32_16x16x32_bf16(                    \
                    false, (A1).v, false, (B0).v, (short)0, acc10, false, false); \
        acc11 = __builtin_amdgcn_wmma_f32_16x16x32_bf16(                    \
                    false, (A1).v, false, (B1).v, (short)0, acc11, false, false); \
    } while (0)

    LOAD_STAGE(a0c, a1c, b0c, b1c, 0);
    int kc = 0;
    for (;;) {
        int kn = kc + 32;
        if (kn < K) LOAD_STAGE(a0n, a1n, b0n, b1n, kn);
        WMMA_STAGE(a0c, a1c, b0c, b1c);
        if (kn >= K) break;
        kc = kn; kn = kc + 32;
        if (kn < K) LOAD_STAGE(a0c, a1c, b0c, b1c, kn);
        WMMA_STAGE(a0n, a1n, b0n, b1n);
        if (kn >= K) break;
        kc = kn;
    }
#undef LOAD_STAGE
#undef WMMA_STAGE

    int n0 = nt * 32 + l, n1 = n0 + 16;
    float sc0 = scale ? scale[n0] : 1.0f, sc1 = scale ? scale[n1] : 1.0f;
    float bs0 = bias  ? bias[n0]  : 0.0f, bs1 = bias  ? bias[n1]  : 0.0f;

#define EPILOG(ACC, MBASE, NCOL, SC, BS, ENABLE)                            \
    if (ENABLE) {                                                           \
        _Pragma("unroll")                                                   \
        for (int r = 0; r < 8; ++r) {                                       \
            int m = (MBASE) + r + half * 8;                                 \
            float v = (ACC)[r] * (SC) + (BS);                               \
            if (act == 1)      v = v > 0.f ? v : 0.f;                       \
            else if (act == 2) v = v > 0.f ? v : 0.01f * v;                 \
            size_t o = (size_t)m * ldo + coloff + (NCOL);                   \
            if (outbf) ((u16*)out)[o] = f2bf(v);                            \
            else       ((float*)out)[o] = v;                                \
        }                                                                   \
    }

    EPILOG(acc00, mt * 32,      n0, sc0, bs0, true);
    EPILOG(acc01, mt * 32,      n1, sc1, bs1, true);
    EPILOG(acc10, mt * 32 + 16, n0, sc0, bs0, hasM1);
    EPILOG(acc11, mt * 32 + 16, n1, sc1, bs1, hasM1);
#undef EPILOG
}

// ---------- max pool over the 32 samples of each center ----------
__global__ void k_maxpool(const u16* __restrict__ Y, int C, int ncent,
                          u16* __restrict__ out) {
    int t = blockIdx.x * blockDim.x + threadIdx.x;
    if (t >= ncent * C) return;
    int c = t % C, ct = t / C;
    const u16* y = Y + (size_t)ct * 32 * C + c;
    float m = -3.4e38f;
    for (int s = 0; s < 32; ++s) {
        float v = bf2f(y[(size_t)s * C]);
        m = v > m ? v : m;
    }
    out[(size_t)ct * C + c] = f2bf(m);
}

// ---------- FP: 3-NN inverse-distance interp + concat skip -> bf16 rows ----------
__global__ void k_interp(const float* __restrict__ uxyz, int ustride, int Nu,
                         const float* __restrict__ kxyz, int Nk,
                         const u16* __restrict__ kfeat, int Ck,
                         const u16* __restrict__ skip, long long skipBStride, int Cs,
                         int Kp, u16* __restrict__ X) {
    int t = blockIdx.x * blockDim.x + threadIdx.x;
    if (t >= 16 * Nu) return;
    int b = t / Nu, u = t % Nu;
    const float* up = uxyz + ((size_t)b * ustride + u) * 3;
    const float* kp = kxyz + (size_t)b * Nk * 3;
    float ux = up[0], uy = up[1], uz = up[2];
    float d0 = 3.4e38f, d1 = 3.4e38f, d2v = 3.4e38f;
    int   i0 = 0, i1 = 0, i2 = 0;
    for (int i = 0; i < Nk; ++i) {
        float dx = kp[i*3]-ux, dy = kp[i*3+1]-uy, dz = kp[i*3+2]-uz;
        float d = dx*dx + dy*dy + dz*dz;
        if (d < d0)       { d2v = d1; i2 = i1; d1 = d0; i1 = i0; d0 = d; i0 = i; }
        else if (d < d1)  { d2v = d1; i2 = i1; d1 = d;  i1 = i; }
        else if (d < d2v) { d2v = d;  i2 = i; }
    }
    float w0 = 1.f/(d0+1e-8f), w1 = 1.f/(d1+1e-8f), w2 = 1.f/(d2v+1e-8f);
    float ws = w0 + w1 + w2; w0 /= ws; w1 /= ws; w2 /= ws;
    const u16* f0 = kfeat + ((size_t)b * Nk + i0) * Ck;
    const u16* f1 = kfeat + ((size_t)b * Nk + i1) * Ck;
    const u16* f2 = kfeat + ((size_t)b * Nk + i2) * Ck;
    u16* xr = X + (size_t)t * Kp;
    for (int c = 0; c < Ck; ++c)
        xr[c] = f2bf(w0*bf2f(f0[c]) + w1*bf2f(f1[c]) + w2*bf2f(f2[c]));
    const u16* sp = skip + (size_t)b * skipBStride + (size_t)u * Cs;
    for (int c = 0; c < Cs; ++c) xr[Ck + c] = sp[c];
    for (int c = Ck + Cs; c < Kp; ++c) xr[c] = 0;
}

// ---------- final head dot product ----------
__global__ void k_final(const u16* __restrict__ net, const float* __restrict__ w2,
                        const float* __restrict__ b2, float* __restrict__ out) {
    int t = threadIdx.x;
    if (t >= 16) return;
    const u16* r = net + t * 128;
    float s = 0.f;
    for (int c = 0; c < 128; ++c) s += bf2f(r[c]) * w2[c];
    out[t] = s + b2[0];
}

// =====================================================================
extern "C" void kernel_launch(void* const* d_in, const int* in_sizes, int n_in,
                              void* d_out, int out_size, void* d_ws, size_t ws_size,
                              hipStream_t stream) {
    (void)in_sizes; (void)n_in; (void)out_size; (void)ws_size;
    const int B = 16, N0 = 4096;
    const float* pcs    = (const float*)d_in[0];
    const float* hidden = (const float*)d_in[1];

    size_t off = 0;
    auto wsAlloc = [&](size_t bytes) -> void* {
        size_t a = (off + 255) & ~(size_t)255;
        off = a + bytes;
        return (void*)((char*)d_ws + a);
    };
    auto pad32 = [](int k) { return (k + 31) & ~31; };

    struct WL { const float* s; const float* b; int N, K, Kp; u16* wb; };
    auto mkw = [&](int wi, int Nn, int Kk) -> WL {
        WL L; L.N = Nn; L.K = Kk; L.Kp = pad32(Kk); L.s = nullptr; L.b = nullptr;
        L.wb = (u16*)wsAlloc((size_t)Nn * L.Kp * sizeof(u16));
        int tot = Nn * L.Kp;
        k_convw<<<(tot + 255) / 256, 256, 0, stream>>>((const float*)d_in[wi], L.wb, Nn, Kk, L.Kp);
        return L;
    };
    auto mksa = [&](int wi, int Nn, int Kk) -> WL {
        WL L = mkw(wi, Nn, Kk);
        L.s = (const float*)d_in[wi + 1];
        L.b = (const float*)d_in[wi + 2];
        return L;
    };
    auto gemm = [&](const u16* X, int ldx, const WL& L, int M, int act,
                    void* out, int ldo, int outbf, int coloff) {
        long tiles = (long)((M + 31) / 32) * (L.N / 32);
        int blocks = (int)((tiles + 7) / 8);
        k_gemm<<<blocks, 256, 0, stream>>>(X, ldx, L.wb, L.Kp, M, L.N, L.Kp,
                                           L.s, L.b, act, out, ldo, outbf, coloff);
    };

    // -------- weight prep (bf16, K zero-padded to multiple of 32) --------
    const int saMid[4] = {32, 64, 128, 256};
    const int saOut[4] = {64, 128, 256, 512};
    const int saCin[4] = {6, 67, 131, 259};
    WL sa[4][3];
    for (int L = 0; L < 4; ++L) {
        int base = 2 + L * 9;
        sa[L][0] = mksa(base + 0, saMid[L], saCin[L]);
        sa[L][1] = mksa(base + 3, saMid[L], saMid[L]);
        sa[L][2] = mksa(base + 6, saOut[L], saMid[L]);
    }
    WL fp0[3], fp1[2], fp2[2], fp3[2];
    fp0[0] = mksa(38, 128, 131); fp0[1] = mksa(41, 128, 128); fp0[2] = mksa(44, 128, 128);
    fp1[0] = mksa(47, 256, 320); fp1[1] = mksa(50, 128, 256);
    fp2[0] = mksa(53, 256, 384); fp2[1] = mksa(56, 256, 256);
    fp3[0] = mksa(59, 256, 768); fp3[1] = mksa(62, 256, 256);
    WL fc = mksa(65, 128, 128);
    WL w3 = mkw(68, 128, 128); w3.b = (const float*)d_in[69];
    WL w1 = mkw(70, 128, 256); w1.b = (const float*)d_in[71];
    const float* w2 = (const float*)d_in[72];
    const float* b2 = (const float*)d_in[73];

    // -------- activation buffers --------
    u16* feats0 = (u16*)wsAlloc((size_t)B * N0 * 3 * 2);
    k_cvt<<<(B * N0 * 3 + 255) / 256, 256, 0, stream>>>(pcs, feats0, B * N0 * 3);
    u16* hiddenb = (u16*)wsAlloc((size_t)B * 128 * 2);
    k_cvt<<<(B * 128 + 255) / 256, 256, 0, stream>>>(hidden, hiddenb, B * 128);

    const int nin[5] = {4096, 1024, 256, 64, 16};
    float* xyzL[5]; xyzL[0] = (float*)pcs;
    for (int L = 1; L < 5; ++L) xyzL[L] = (float*)wsAlloc((size_t)B * nin[L] * 3 * 4);
    int* fidx = (int*)wsAlloc((size_t)B * 1024 * 4);
    int* bq   = (int*)wsAlloc((size_t)B * 1024 * 32 * 4);
    u16* pooled[5]; pooled[0] = feats0;
    for (int L = 1; L < 5; ++L) pooled[L] = (u16*)wsAlloc((size_t)B * nin[L] * saOut[L - 1] * 2);
    const size_t pmax = (size_t)B * 1024 * 32 * 64 * 2;  // 64 MB (SA0 layer-3 output)
    u16* bufA = (u16*)wsAlloc(pmax);
    u16* bufB = (u16*)wsAlloc(pmax);

    // -------- SA levels --------
    const float radsq[4] = {0.01f, 0.04f, 0.16f, 0.64f};
    const int   npt[4]   = {1024, 256, 64, 16};
    int CfIn = 3;
    for (int L = 0; L < 4; ++L) {
        int n = nin[L], np = npt[L], nc = B * np;
        k_fps<<<B, 256, 0, stream>>>(xyzL[L], n, np, fidx, xyzL[L + 1]);
        k_bq<<<(nc + 255) / 256, 256, 0, stream>>>(xyzL[L], xyzL[L + 1], n, np, radsq[L], bq);
        int rows = nc * 32;
        int Kp0 = pad32(3 + CfIn);
        k_group<<<(rows + 255) / 256, 256, 0, stream>>>(xyzL[L], xyzL[L + 1], pooled[L],
                                                        CfIn, bq, n, np, Kp0, bufA);
        gemm(bufA, Kp0,      sa[L][0], rows, 1, bufB, saMid[L], 1, 0);
        gemm(bufB, saMid[L], sa[L][1], rows, 1, bufA, saMid[L], 1, 0);
        gemm(bufA, saMid[L], sa[L][2], rows, 1, bufB, saOut[L], 1, 0);
        k_maxpool<<<(nc * saOut[L] + 255) / 256, 256, 0, stream>>>(bufB, saOut[L], nc, pooled[L + 1]);
        CfIn = saOut[L];
    }

    // -------- FP levels (j = 3,2,1, then pruned j = 0) --------
    u16* fpoutb[4];
    fpoutb[3] = (u16*)wsAlloc((size_t)B * 64 * 256 * 2);
    fpoutb[2] = (u16*)wsAlloc((size_t)B * 256 * 256 * 2);
    fpoutb[1] = (u16*)wsAlloc((size_t)B * 1024 * 128 * 2);
    fpoutb[0] = (u16*)wsAlloc((size_t)B * 128 * 2);

    { // FP3: 64 unknown, 16 known, interp 512 + skip 256 -> [768,256,256]
        int Nu = 64, Nk = 16, Ck = 512, Cs = 256, Kp = 768, rows = B * Nu;
        k_interp<<<(rows + 255) / 256, 256, 0, stream>>>(xyzL[3], Nu, Nu, xyzL[4], Nk,
            pooled[4], Ck, pooled[3], (long long)Nu * Cs, Cs, Kp, bufA);
        gemm(bufA, Kp, fp3[0], rows, 1, bufB, 256, 1, 0);
        gemm(bufB, 256, fp3[1], rows, 1, fpoutb[3], 256, 1, 0);
    }
    { // FP2: 256 unknown, 64 known, interp 256 + skip 128 -> [384,256,256]
        int Nu = 256, Nk = 64, Ck = 256, Cs = 128, Kp = 384, rows = B * Nu;
        k_interp<<<(rows + 255) / 256, 256, 0, stream>>>(xyzL[2], Nu, Nu, xyzL[3], Nk,
            fpoutb[3], Ck, pooled[2], (long long)Nu * Cs, Cs, Kp, bufA);
        gemm(bufA, Kp, fp2[0], rows, 1, bufB, 256, 1, 0);
        gemm(bufB, 256, fp2[1], rows, 1, fpoutb[2], 256, 1, 0);
    }
    { // FP1: 1024 unknown, 256 known, interp 256 + skip 64 -> [320,256,128]
        int Nu = 1024, Nk = 256, Ck = 256, Cs = 64, Kp = 320, rows = B * Nu;
        k_interp<<<(rows + 255) / 256, 256, 0, stream>>>(xyzL[1], Nu, Nu, xyzL[2], Nk,
            fpoutb[2], Ck, pooled[1], (long long)Nu * Cs, Cs, Kp, bufA);
        gemm(bufA, Kp, fp1[0], rows, 1, bufB, 256, 1, 0);
        gemm(bufB, 256, fp1[1], rows, 1, fpoutb[1], 128, 1, 0);
    }
    { // FP0: ONLY point 0 per batch feeds the head -> 16 rows total
        int Nk = 1024, Ck = 128, Cs = 3, Kp = 160, rows = B;
        k_interp<<<1, 256, 0, stream>>>(pcs, 4096, 1, xyzL[1], Nk,
            fpoutb[1], Ck, feats0, (long long)4096 * 3, Cs, Kp, bufA);
        gemm(bufA, Kp, fp0[0], rows, 1, bufB, 128, 1, 0);
        gemm(bufB, 128, fp0[1], rows, 1, bufA, 128, 1, 0);
        gemm(bufA, 128, fp0[2], rows, 1, fpoutb[0], 128, 1, 0);
    }

    // -------- head --------
    u16* concatb = (u16*)wsAlloc((size_t)16 * 256 * 2);
    u16* net1    = (u16*)wsAlloc((size_t)16 * 128 * 2);
    gemm(fpoutb[0], 128, fc, 16, 1, concatb, 256, 1, 0);    // FC(point0) -> cols [0,128)
    gemm(hiddenb,   128, w3, 16, 0, concatb, 256, 1, 128);  // hidden @ w3^T + b3 -> cols [128,256)
    gemm(concatb,   256, w1, 16, 2, net1, 128, 1, 0);       // leaky_relu
    k_final<<<1, 16, 0, stream>>>(net1, w2, b2, (float*)d_out);
}
